// GaussianMixtureModel_18537078850250
// MI455X (gfx1250) — matrix operand order repaired and checked by reading, not compiled, not run on hardware
//
#include <hip/hip_runtime.h>
#include <math.h>

typedef __attribute__((ext_vector_type(8)))  float  v8f;
typedef __attribute__((ext_vector_type(16))) __bf16 v16bf;
typedef __attribute__((ext_vector_type(4)))  int    v4i;

#define DFEAT 128
#define KCOMP 16
#define BATCH 65536

// ---------------------------------------------------------------------------
// Pre-kernel A: split P into bf16 hi/lo planes (bf16x3 trick):
//   hi = bf16(p), lo = bf16(p - hi);  p*q ~= hi*hi + hi*lo + lo*hi (fp32 acc)
// ---------------------------------------------------------------------------
__global__ __launch_bounds__(256)
void gmm_cvt(const float* __restrict__ P, __bf16* __restrict__ Phi,
             __bf16* __restrict__ Plo) {
    const int idx = blockIdx.x * 256 + threadIdx.x;   // 262144 elements
    const float p = P[idx];
    const __bf16 h = (__bf16)p;
    Phi[idx] = h;
    Plo[idx] = (__bf16)(p - (float)h);
}

// ---------------------------------------------------------------------------
// Pre-kernel B: z[k][i] = (P_k m_k)[i];  ck[k] = log w_k + logdet_k - 64*log2pi
// ---------------------------------------------------------------------------
__global__ __launch_bounds__(DFEAT)
void gmm_pre(const float* __restrict__ P, const float* __restrict__ means,
             const float* __restrict__ logw, float* __restrict__ z,
             float* __restrict__ ck) {
    const int k = blockIdx.x;
    const int i = threadIdx.x;
    const float* Prow = P + ((size_t)k * DFEAT + i) * DFEAT;
    const float* mk   = means + (size_t)k * DFEAT;
    float acc = 0.0f;
#pragma unroll 8
    for (int j = 0; j < DFEAT; ++j) acc = fmaf(Prow[j], mk[j], acc);
    z[k * DFEAT + i] = acc;

    __shared__ float red[DFEAT];
    red[i] = logf(Prow[i]);
    __syncthreads();
    for (int s2 = DFEAT / 2; s2 > 0; s2 >>= 1) {
        if (i < s2) red[i] += red[i + s2];
        __syncthreads();
    }
    if (i == 0) {
        const float log2pi = 1.8378770664093453f;
        ck[k] = logw[k] + red[0] - 0.5f * (float)DFEAT * log2pi;
    }
}

// ---------------------------------------------------------------------------
// Async staging helpers: global -> LDS, 16 bytes per call per thread.
// Probe-derived signature: arg0 = v4i addrspace(1)* (non-const, global src).
// ---------------------------------------------------------------------------
__device__ __forceinline__ void stage_copy16(const char* g, char* l) {
#if __has_builtin(__builtin_amdgcn_global_load_async_to_lds_b128)
    __builtin_amdgcn_global_load_async_to_lds_b128(
        (__attribute__((address_space(1))) v4i*)g,
        (__attribute__((address_space(3))) v4i*)l, 0, 0);
#else
    *reinterpret_cast<uint4*>(l) = *reinterpret_cast<const uint4*>(g);
#endif
}

__device__ __forceinline__ void stage_wait() {
#if __has_builtin(__builtin_amdgcn_global_load_async_to_lds_b128)
    asm volatile("s_wait_asynccnt 0x0" ::: "memory");
#endif
}

// ---------------------------------------------------------------------------
// Main kernel: one wave32 -> 16 batch rows; block = 8 waves -> 128 rows.
// U = X * P_k^T via bf16x3 on v_wmma_f32_16x16x32_bf16; P_k staged in LDS
// (double-buffered async copies); quad-form + online logsumexp in registers.
// ---------------------------------------------------------------------------
__global__ __launch_bounds__(256)
void gmm_main(const float* __restrict__ x,
              const __bf16* __restrict__ Phi, const __bf16* __restrict__ Plo,
              const float* __restrict__ z, const float* __restrict__ ck,
              float* __restrict__ out) {
    const int tid  = threadIdx.x;
    const int lane = tid & 31;
    const int wave = tid >> 5;
    const int half = lane >> 4;     // 0: lanes 0-15, 1: lanes 16-31
    const int l16  = lane & 15;
    const int b0   = blockIdx.x * 128 + wave * 16;

    // LDS: double-buffered P_k planes. Each buffer: Phi_k [0,32KB) Plo_k [32KB,64KB)
    __shared__ __align__(64) char ldsbuf[2][65536];

    // ---- A fragments: decompose x tile into bf16 hi/lo, resident all k ----
    // A(16x32 bf16) layout: lane row M=l16; elems 0-15 <-> K=16h+0..15
    //                       (within 32-wide chunk)
    v16bf ah[4], al[4];
    {
        const float* xrow = x + (size_t)(b0 + l16) * DFEAT;
#pragma unroll
        for (int q = 0; q < 4; ++q) {
            const float* g0 = xrow + 32 * q + 16 * half;
            const float* g1 = xrow + 32 * q + 16 * half + 8;
#pragma unroll
            for (int e = 0; e < 8; ++e) {
                const float f0 = g0[e];
                const __bf16 h0 = (__bf16)f0;
                ah[q][e] = h0;
                al[q][e] = (__bf16)(f0 - (float)h0);
                const float f1 = g1[e];
                const __bf16 h1 = (__bf16)f1;
                ah[q][8 + e] = h1;
                al[q][8 + e] = (__bf16)(f1 - (float)h1);
            }
        }
    }

    float m[8], sacc[8];
#pragma unroll
    for (int r = 0; r < 8; ++r) { m[r] = -INFINITY; sacc[r] = 0.0f; }

    // ---- stage component 0 ----
    {
        const char* gh = (const char*)Phi + tid * 16;
        const char* gl = (const char*)Plo + tid * 16;
        char* lh = ldsbuf[0] + tid * 16;
        char* ll = ldsbuf[0] + 32768 + tid * 16;
#pragma unroll
        for (int r = 0; r < 8; ++r) {
            stage_copy16(gh + r * 4096, lh + r * 4096);
            stage_copy16(gl + r * 4096, ll + r * 4096);
        }
    }
    stage_wait();
    __syncthreads();

    for (int k = 0; k < KCOMP; ++k) {
        const int cur = k & 1;
        // prefetch next component into the other buffer (overlaps compute)
        if (k + 1 < KCOMP) {
            const char* gh = (const char*)Phi + (size_t)(k + 1) * 32768 + tid * 16;
            const char* gl = (const char*)Plo + (size_t)(k + 1) * 32768 + tid * 16;
            char* lh = ldsbuf[cur ^ 1] + tid * 16;
            char* ll = ldsbuf[cur ^ 1] + 32768 + tid * 16;
#pragma unroll
            for (int r = 0; r < 8; ++r) {
                stage_copy16(gh + r * 4096, lh + r * 4096);
                stage_copy16(gl + r * 4096, ll + r * 4096);
            }
        }

        const __bf16* bufPhi = reinterpret_cast<const __bf16*>(ldsbuf[cur]);
        const __bf16* bufPlo = bufPhi + 16384;
        const float ckv = ck[k];

        float s[8];
#pragma unroll
        for (int r = 0; r < 8; ++r) s[r] = 0.0f;

        for (int it = 0; it < 8; ++it) {          // i-tile: i0 .. i0+15
            const int i0 = it * 16;
            v8f c0 = {0.f,0.f,0.f,0.f,0.f,0.f,0.f,0.f};   // hi*hi chain
            v8f c1 = {0.f,0.f,0.f,0.f,0.f,0.f,0.f,0.f};   // cross-term chain
            // B(32x16 bf16) layout: lane col N=l16, elems 0-15 <-> K=16h+0..15
            // -> 16 contiguous bf16 from row (i0+l16) at col 32q+16h.
            const int rowoff = (i0 + l16) * DFEAT + 16 * half;
#pragma unroll
            for (int q = 0; q < 4; ++q) {
                const v16bf bh = *reinterpret_cast<const v16bf*>(bufPhi + rowoff + 32 * q);
                const v16bf bl = *reinterpret_cast<const v16bf*>(bufPlo + rowoff + 32 * q);
                c0 = __builtin_amdgcn_wmma_f32_16x16x32_bf16(
                         false, ah[q], false, bh, (short)0, c0, false, false);
                c1 = __builtin_amdgcn_wmma_f32_16x16x32_bf16(
                         false, ah[q], false, bl, (short)0, c1, false, false);
                c1 = __builtin_amdgcn_wmma_f32_16x16x32_bf16(
                         false, al[q], false, bh, (short)0, c1, false, false);
            }
            // C layout: row M = r + 8*half, col N = i0 + l16
            const float zv = z[k * DFEAT + i0 + l16];
#pragma unroll
            for (int r = 0; r < 8; ++r) {
                const float d = (c0[r] + c1[r]) - zv;
                s[r] = fmaf(d, d, s[r]);
            }
        }

        // reduce quad-form over the 16 lanes of each half-group
#pragma unroll
        for (int r = 0; r < 8; ++r) {
            float v = s[r];
            v += __shfl_xor(v, 1, 32);
            v += __shfl_xor(v, 2, 32);
            v += __shfl_xor(v, 4, 32);
            v += __shfl_xor(v, 8, 32);
            s[r] = v;
        }

        // online logsumexp
#pragma unroll
        for (int r = 0; r < 8; ++r) {
            const float L  = fmaf(-0.5f, s[r], ckv);
            const float mo = m[r];
            const float mn = fmaxf(mo, L);
            sacc[r] = sacc[r] * expf(mo - mn) + expf(L - mn);
            m[r] = mn;
        }

        stage_wait();        // next buffer's async copies complete (this wave)
        __syncthreads();     // all waves done staging + done reading cur
    }

    if (l16 == 0) {
#pragma unroll
        for (int r = 0; r < 8; ++r)
            out[b0 + 8 * half + r] = m[r] + logf(sacc[r]);
    }
}

// ---------------------------------------------------------------------------
extern "C" void kernel_launch(void* const* d_in, const int* in_sizes, int n_in,
                              void* d_out, int out_size, void* d_ws, size_t ws_size,
                              hipStream_t stream) {
    (void)in_sizes; (void)n_in; (void)out_size; (void)ws_size;
    const float* x     = (const float*)d_in[0];  // [65536,128]
    const float* means = (const float*)d_in[1];  // [16,128]
    const float* P     = (const float*)d_in[2];  // [16,128,128]
    const float* logw  = (const float*)d_in[3];  // [16]
    float* out = (float*)d_out;                  // [65536]

    // workspace layout
    float*  z   = (float*)d_ws;                              // 2048 floats
    float*  ck  = z + KCOMP * DFEAT;                         // 16 floats
    __bf16* Phi = (__bf16*)((char*)d_ws + 16384);            // 512 KB
    __bf16* Plo = Phi + (size_t)KCOMP * DFEAT * DFEAT;       // 512 KB

    gmm_cvt<<<(KCOMP * DFEAT * DFEAT) / 256, 256, 0, stream>>>(P, Phi, Plo);
    gmm_pre<<<KCOMP, DFEAT, 0, stream>>>(P, means, logw, z, ck);
    gmm_main<<<BATCH / 128, 256, 0, stream>>>(x, Phi, Plo, z, ck, out);
}
